// MaxViTBlock_83708912599204
// MI455X (gfx1250) — matrix operand order repaired and checked by reading, not compiled, not run on hardware
//
#include <hip/hip_runtime.h>
#include <hip/hip_bf16.h>

// ---------------- types / helpers ----------------
typedef __attribute__((ext_vector_type(16))) __bf16       v16bf;
typedef __attribute__((ext_vector_type(8)))  float        v8f;
typedef __attribute__((ext_vector_type(8)))  unsigned int v8u;

#define DEV __device__ __forceinline__

DEV unsigned short f2bf(float f) {            // f32 -> bf16, round-to-nearest-even
  unsigned int u = __builtin_bit_cast(unsigned int, f);
  u += 0x7fffu + ((u >> 16) & 1u);
  return (unsigned short)(u >> 16);
}
DEV float bf2f(unsigned short h) {
  unsigned int u = ((unsigned int)h) << 16;
  return __builtin_bit_cast(float, u);
}
DEV float gelu_erf(float x) { return 0.5f * x * (1.0f + erff(x * 0.70710678118654752440f)); }

DEV v8f wmma_bf16(v16bf a, v16bf b, v8f c) {
  return __builtin_amdgcn_wmma_f32_16x16x32_bf16(false, a, false, b, (short)0, c, false, false);
}

// dims
#define BSZ   16
#define CINC  96
#define COUTC 192
#define NHEAD 6
#define P0    (16*112*112)   // 200704 pixels @112
#define P1    (16*56*56)     // 50176  pixels @56

// window/grid token -> pixel index (channel-last [b,h,w,:] over 56x56)
DEV int token_pixel(int b_, int n, int grid_mode) {
  int b  = b_ >> 6;
  int hh = (b_ & 63) >> 3;
  int ww = b_ & 7;
  int i = n / 7, j = n % 7;
  int h, w;
  if (grid_mode) { h = i * 8 + hh; w = j * 8 + ww; }   // grid partition
  else           { h = hh * 7 + i; w = ww * 7 + j; }   // window partition
  return (b * 56 + h) * 56 + w;
}

// ---------------- param prep (BN fold, q-scale fold) ----------------
__global__ void prep_params(
    const float* bn0_g, const float* bn0_b, const float* bn0_m, const float* bn0_v,
    const float* dw_b,  const float* dwbn_g, const float* dwbn_b, const float* dwbn_m, const float* dwbn_v,
    const float* pw_b,  const float* pwbn_g, const float* pwbn_b, const float* pwbn_m, const float* pwbn_v,
    const float* qkvb_w, const float* qkvb_g,
    float* scale0, float* shift0, float* sdw, float* tdw, float* spw, float* tpw,
    float* qscW, float* qshW, float* qscG, float* qshG)
{
  int t = threadIdx.x;
  if (t < 96) {
    float i0 = rsqrtf(bn0_v[t] + 1e-5f);
    float s0 = bn0_g[t] * i0;
    scale0[t] = s0; shift0[t] = bn0_b[t] - bn0_m[t] * s0;
    float i1 = rsqrtf(dwbn_v[t] + 1e-5f);
    float s1 = dwbn_g[t] * i1;
    sdw[t] = s1; tdw[t] = (dw_b[t] - dwbn_m[t]) * s1 + dwbn_b[t];
  }
  if (t < 192) {
    float i2 = rsqrtf(pwbn_v[t] + 1e-5f);
    float s2 = pwbn_g[t] * i2;
    spw[t] = s2; tpw[t] = (pw_b[t] - pwbn_m[t]) * s2 + pwbn_b[t];
  }
  if (t < 576) {
    float s = (t < 192) ? 0.40824829046386301637f : 1.0f;  // HEADS^-0.5 applied to q
    qscW[t] = s; qshW[t] = qkvb_w[t] * s;
    qscG[t] = s; qshG[t] = qkvb_g[t] * s;
  }
}

// ---------------- simple converters ----------------
// converts n elements; zero-fills [n, npad) so GEMM weight tiles need no N guard
__global__ void cvt_f2bf(const float* __restrict__ in, unsigned short* __restrict__ out,
                         int n, int npad) {
  int i = blockIdx.x * 256 + threadIdx.x;
  if (i < n) out[i] = f2bf(in[i]);
  else if (i < npad) out[i] = 0;
}

// BN0 + NCHW(112) -> channel-last bf16
__global__ void cvt0_kernel(const float* __restrict__ x, const float* __restrict__ sc,
                            const float* __restrict__ sh, unsigned short* __restrict__ out) {
  int idx = blockIdx.x * 256 + threadIdx.x;
  if (idx >= P0 * CINC) return;
  int c = idx % CINC; int pix = idx / CINC;
  int w = pix % 112, h = (pix / 112) % 112, b = pix / (112 * 112);
  float v = x[(((size_t)b * CINC + c) * 112 + h) * 112 + w];
  out[idx] = f2bf(v * sc[c] + sh[c]);
}

// maxpool 2x2 on x (NCHW 112) -> channel-last bf16 [P1,96]
__global__ void maxpool_cvt(const float* __restrict__ x, unsigned short* __restrict__ out) {
  int idx = blockIdx.x * 256 + threadIdx.x;
  if (idx >= P1 * CINC) return;
  int c = idx % CINC; int pix = idx / CINC;
  int ow = pix % 56, oh = (pix / 56) % 56, b = pix / (56 * 56);
  size_t base = (((size_t)b * CINC + c) * 112 + oh * 2) * 112 + ow * 2;
  float m = fmaxf(fmaxf(x[base], x[base + 1]), fmaxf(x[base + 112], x[base + 113]));
  out[idx] = f2bf(m);
}

// channel-last f32 -> NCHW f32 output
__global__ void out_transpose(const float* __restrict__ in, float* __restrict__ out) {
  int idx = blockIdx.x * 256 + threadIdx.x;
  if (idx >= P1 * COUTC) return;
  int w = idx % 56, h = (idx / 56) % 56, c = (idx / 3136) % COUTC, b = idx / (3136 * COUTC);
  out[idx] = in[((size_t)b * 3136 + h * 56 + w) * COUTC + c];
}

// ---------------- depthwise 3x3 stride-2 + BN + GELU ----------------
__global__ void dwconv_kernel(const unsigned short* __restrict__ in, const float* __restrict__ w9,
                              const float* __restrict__ sdw, const float* __restrict__ tdw,
                              unsigned short* __restrict__ out) {
  int idx = blockIdx.x * 256 + threadIdx.x;
  if (idx >= P1 * CINC) return;
  int c = idx % CINC; int pix = idx / CINC;
  int ow = pix % 56, oh = (pix / 56) % 56, b = pix / (56 * 56);
  float acc = 0.f;
  #pragma unroll
  for (int dy = 0; dy < 3; ++dy) {
    int ih = oh * 2 - 1 + dy;
    if (ih < 0 || ih >= 112) continue;
    #pragma unroll
    for (int dx = 0; dx < 3; ++dx) {
      int iw = ow * 2 - 1 + dx;
      if (iw < 0 || iw >= 112) continue;
      acc += bf2f(in[(((size_t)b * 112 + ih) * 112 + iw) * CINC + c]) * w9[c * 9 + dy * 3 + dx];
    }
  }
  out[idx] = f2bf(gelu_erf(acc * sdw[c] + tdw[c]));
}

// ---------------- squeeze-excitation (two-stage spatial mean) ----------------
__global__ void se_pool_partial(const unsigned short* __restrict__ h3, float* __restrict__ partial) {
  int b = blockIdx.x, chunk = blockIdx.y, c = threadIdx.x;   // grid (16,49), 192 threads
  const unsigned short* p = h3 + ((size_t)b * 3136 + chunk * 64) * COUTC + c;
  float s = 0.f;
  for (int i = 0; i < 64; ++i) s += bf2f(p[(size_t)i * COUTC]);
  partial[(b * 49 + chunk) * COUTC + c] = s;
}

__global__ void se_pool_final(const float* __restrict__ partial, float* __restrict__ pool) {
  int b = blockIdx.x, c = threadIdx.x;                       // grid 16, 192 threads
  float s = 0.f;
  for (int k = 0; k < 49; ++k) s += partial[(b * 49 + k) * COUTC + c];
  pool[b * COUTC + c] = s * (1.0f / 3136.0f);
}

__global__ void se_fc(const float* __restrict__ pool,
                      const float* __restrict__ w1, const float* __restrict__ b1,
                      const float* __restrict__ w2, const float* __restrict__ b2,
                      float* __restrict__ sig) {
  __shared__ float mid[48];
  int b = blockIdx.x, t = threadIdx.x;                 // 192 threads
  if (t < 48) {
    float s = b1[t];
    for (int c = 0; c < COUTC; ++c) s += pool[b * COUTC + c] * w1[t * COUTC + c];
    mid[t] = gelu_erf(s);
  }
  __syncthreads();
  float s = b2[t];
  for (int r = 0; r < 48; ++r) s += mid[r] * w2[t * 48 + r];
  sig[b * COUTC + t] = 1.0f / (1.0f + expf(-s));
}

__global__ void se_scale(const unsigned short* __restrict__ h3, const float* __restrict__ sig,
                         unsigned short* __restrict__ out) {
  int idx = blockIdx.x * 256 + threadIdx.x;
  if (idx >= P1 * COUTC) return;
  int c = idx % COUTC; int b = (idx / COUTC) / 3136;
  out[idx] = f2bf(bf2f(h3[idx]) * sig[b * COUTC + c]);
}

// ---------------- LayerNorm (per 192-ch row) -> bf16 ----------------
__global__ __launch_bounds__(128) void ln_kernel(const float* __restrict__ x,
                                                 const float* __restrict__ g, const float* __restrict__ bta,
                                                 unsigned short* __restrict__ out) {
  int wave = threadIdx.x >> 5, lane = threadIdx.x & 31;
  int tok = blockIdx.x * 4 + wave;
  if (tok >= P1) return;
  const float* row = x + (size_t)tok * COUTC;
  float v[6]; float s = 0.f;
  #pragma unroll
  for (int i = 0; i < 6; ++i) { v[i] = row[lane + 32 * i]; s += v[i]; }
  #pragma unroll
  for (int o = 16; o; o >>= 1) s += __shfl_xor(s, o, 32);
  float mean = s * (1.0f / 192.0f);
  float q = 0.f;
  #pragma unroll
  for (int i = 0; i < 6; ++i) { float d = v[i] - mean; q += d * d; }
  #pragma unroll
  for (int o = 16; o; o >>= 1) q += __shfl_xor(q, o, 32);
  float inv = rsqrtf(q * (1.0f / 192.0f) + 1e-5f);
  #pragma unroll
  for (int i = 0; i < 6; ++i) {
    int c = lane + 32 * i;
    out[(size_t)tok * COUTC + c] = f2bf((v[i] - mean) * inv * g[c] + bta[c]);
  }
}

// ---------------- generic bf16 WMMA GEMM (double-buffered) ----------------
// C[M,N] = ((A[M,K] @ W[N,K]^T) * scale[n] + shift[n]) |> optional GELU |> + addend
// 128x64 block tile, 8 waves x four 16x16 WMMA tiles, K staged 32-wide through
// double-buffered LDS: global loads for tile k+1 overlap WMMA of tile k,
// one barrier per K-step.
// Requires: M % 128 == 0, K % 32 == 0, W padded with zero rows to 64-multiple.
__global__ __launch_bounds__(256) void gemm_bf16(
    const unsigned short* __restrict__ A, const unsigned short* __restrict__ W,
    const float* __restrict__ scale, const float* __restrict__ shift,
    const float* __restrict__ addend, float* __restrict__ outF,
    unsigned short* __restrict__ outB,
    int Mdim, int Ndim, int Kdim, int gelu_flag)
{
  __shared__ __align__(16) unsigned int lds_a[2][128 * 20];   // row stride 20 dwords
  __shared__ __align__(16) unsigned int lds_w[2][64 * 20];
  const int tid = threadIdx.x, lane = tid & 31, wave = tid >> 5;
  const int hi = lane >> 4, nl = lane & 15;
  const int m0 = blockIdx.x * 128, n0 = blockIdx.y * 64;
  const int nsub = wave & 3, mg = wave >> 2;   // wave owns m-tiles mg+{0,2,4,6}, n-tile nsub

  v8f acc[4];
  #pragma unroll
  for (int t = 0; t < 4; ++t) acc[t] = (v8f){0,0,0,0,0,0,0,0};

  const unsigned int* A32 = (const unsigned int*)A;
  const unsigned int* W32 = (const unsigned int*)W;
  const int kdw = Kdim >> 1;
  const int r0 = tid >> 2, c4 = (tid & 3) * 4;   // staging: thread's row / dword-chunk

  // warm L2 for the residual tile read in the epilogue (gfx1250 global_prefetch_b8)
  if (addend) __builtin_prefetch(&addend[(size_t)(m0 + (tid >> 1)) * Ndim + n0 + (tid & 1) * 32], 0, 0);

  const unsigned int* pa0 = A32 + (size_t)(m0 + r0)      * kdw + c4;
  const unsigned int* pa1 = A32 + (size_t)(m0 + 64 + r0) * kdw + c4;
  const unsigned int* pw  = W32 + (size_t)(n0 + r0)      * kdw + c4;

  uint4 ra0 = *(const uint4*)pa0;     // prologue: load tile 0
  uint4 ra1 = *(const uint4*)pa1;
  uint4 rw  = *(const uint4*)pw;

  const int ksteps = Kdim >> 5;
  for (int ks = 0; ks < ksteps; ++ks) {
    const int cur = ks & 1;
    // commit staged regs into current LDS buffer
    *(uint4*)(&lds_a[cur][r0 * 20 + c4])        = ra0;
    *(uint4*)(&lds_a[cur][(64 + r0) * 20 + c4]) = ra1;
    *(uint4*)(&lds_w[cur][r0 * 20 + c4])        = rw;
    __syncthreads();
    // prefetch tile k+1 from global (retires during WMMA below)
    if (ks + 1 < ksteps) {
      const int koff = (ks + 1) << 4;            // 16 dwords per K-step
      ra0 = *(const uint4*)(pa0 + koff);
      ra1 = *(const uint4*)(pa1 + koff);
      rw  = *(const uint4*)(pw + koff);
    }
    // compute tile k
    v8u bu;
    const int brow = nsub * 16 + nl;
    #pragma unroll
    for (int d = 0; d < 8; ++d) bu[d] = lds_w[cur][brow * 20 + (hi ? 8 : 0) + d];  // contiguous k = 16*hi + s (same as A)
    v16bf bfr = __builtin_bit_cast(v16bf, bu);
    #pragma unroll
    for (int t = 0; t < 4; ++t) {
      const int arow = (mg + 2 * t) * 16 + nl;
      v8u au;
      #pragma unroll
      for (int d = 0; d < 8; ++d) {
        int k2 = (hi ? 8 : 0) + d;                        // contiguous k = 16*hi + s (same as B)
        au[d] = lds_a[cur][arow * 20 + k2];
      }
      acc[t] = wmma_bf16(__builtin_bit_cast(v16bf, au), bfr, acc[t]);
    }
  }

  const int ncol = n0 + nsub * 16 + nl;
  if (ncol >= Ndim) return;
  const float sc = scale ? scale[ncol] : 1.0f;
  const float sh = shift ? shift[ncol] : 0.0f;
  #pragma unroll
  for (int t = 0; t < 4; ++t) {
    const int mbase = m0 + (mg + 2 * t) * 16 + hi * 8;
    #pragma unroll
    for (int r = 0; r < 8; ++r) {
      int mrow = mbase + r;
      float y = acc[t][r] * sc + sh;
      if (gelu_flag) y = gelu_erf(y);
      size_t o = (size_t)mrow * Ndim + ncol;
      if (addend) y += addend[o];
      if (outF) outF[o] = y;
      if (outB) outB[o] = f2bf(y);
    }
  }
}

// ---------------- attention core: per (window, head) ----------------
// S = (q*scale) K^T + relbias ; softmax ; O = P V    (49 tokens padded to 64, d=32)
__global__ __launch_bounds__(128) void attn_core(
    const unsigned short* __restrict__ qkv, const float* __restrict__ tab,
    unsigned short* __restrict__ attn_out, int grid_mode)
{
  __shared__ __align__(16) unsigned int qs[64 * 20];
  __shared__ __align__(16) unsigned int ks_[64 * 20];
  __shared__ __align__(16) unsigned int vt[32 * 36];   // V transposed: [dim][token]
  __shared__ float        Sf[64 * 65];
  __shared__ __align__(16) unsigned int ps[64 * 36];   // softmaxed P, bf16
  __shared__ float        biasL[169];

  const int tid = threadIdx.x;
  const int b_ = blockIdx.x, head = blockIdx.y;

  for (int t = tid; t < 169; t += 128) biasL[t] = tab[t * NHEAD + head];

  for (int idx = tid; idx < 1024; idx += 128) {        // Q, K: 64 tok x 16 dwords
    int tok = idx >> 4, c = idx & 15;
    unsigned int qv = 0, kv = 0;
    if (tok < 49) {
      size_t base = (size_t)token_pixel(b_, tok, grid_mode) * 576 + head * 32;
      const unsigned int* q32 = (const unsigned int*)(qkv + base);
      qv = q32[c];
      kv = q32[96 + c];                                // k at +192 bf16 = +96 dwords
    }
    qs[tok * 20 + c] = qv;
    ks_[tok * 20 + c] = kv;
  }
  unsigned short* vtu = (unsigned short*)vt;
  for (int idx = tid; idx < 2048; idx += 128) {        // V transposed
    int tok = idx >> 5, d = idx & 31;
    unsigned short v = 0;
    if (tok < 49) v = qkv[(size_t)token_pixel(b_, tok, grid_mode) * 576 + 384 + head * 32 + d];
    vtu[d * 72 + tok] = v;
  }
  __syncthreads();

  const int lane = tid & 31, wave = tid >> 5;
  const int hi = lane >> 4, nl = lane & 15;

  { // S = Q K^T  (each wave: 16 rows x 64 cols = 4 WMMA)
    v8u au;
    int arow = wave * 16 + nl;
    #pragma unroll
    for (int d = 0; d < 8; ++d) {
      int k2 = (hi ? 8 : 0) + d;
      au[d] = qs[arow * 20 + k2];
    }
    v16bf af = __builtin_bit_cast(v16bf, au);
    #pragma unroll
    for (int nsubt = 0; nsubt < 4; ++nsubt) {
      v8u bu;
      int brow = nsubt * 16 + nl;
      #pragma unroll
      for (int d = 0; d < 8; ++d) bu[d] = ks_[brow * 20 + (hi ? 8 : 0) + d];
      v8f acc = {0,0,0,0,0,0,0,0};
      acc = wmma_bf16(af, __builtin_bit_cast(v16bf, bu), acc);
      #pragma unroll
      for (int r = 0; r < 8; ++r) {
        int i = wave * 16 + hi * 8 + r;
        int j = nsubt * 16 + nl;
        float val = -1e30f;
        if (i < 49 && j < 49) {
          int dr = i / 7 - j / 7 + 6;
          int dc = i % 7 - j % 7 + 6;
          val = acc[r] + biasL[dr * 13 + dc];
        }
        Sf[i * 65 + j] = val;
      }
    }
  }
  __syncthreads();

  if (tid < 64) {                                      // softmax row tid
    int r = tid;
    unsigned short* psu = (unsigned short*)ps;
    if (r < 49) {
      float mx = Sf[r * 65];
      for (int c = 1; c < 64; ++c) mx = fmaxf(mx, Sf[r * 65 + c]);
      float sum = 0.f;
      for (int c = 0; c < 64; ++c) {
        float e = expf(Sf[r * 65 + c] - mx);
        Sf[r * 65 + c] = e;                            // cache exp for pass 2
        sum += e;
      }
      float inv = 1.0f / sum;
      for (int c = 0; c < 64; ++c) psu[r * 72 + c] = f2bf(Sf[r * 65 + c] * inv);
    } else {
      for (int c = 0; c < 64; ++c) psu[r * 72 + c] = 0;
    }
  }
  __syncthreads();

  v8f o0 = {0,0,0,0,0,0,0,0}, o1 = {0,0,0,0,0,0,0,0};  // O = P V (K=64 -> 2 steps)
  #pragma unroll
  for (int kk = 0; kk < 2; ++kk) {
    v8u au;
    int arow = wave * 16 + nl;
    #pragma unroll
    for (int d = 0; d < 8; ++d) {
      int k2 = (hi ? 8 : 0) + d;
      au[d] = ps[arow * 36 + kk * 16 + k2];
    }
    v16bf af = __builtin_bit_cast(v16bf, au);
    v8u b0, b1;
    #pragma unroll
    for (int d = 0; d < 8; ++d) {
      int o = kk * 16 + (hi ? 8 : 0) + d;
      b0[d] = vt[nl * 36 + o];
      b1[d] = vt[(16 + nl) * 36 + o];
    }
    o0 = wmma_bf16(af, __builtin_bit_cast(v16bf, b0), o0);
    o1 = wmma_bf16(af, __builtin_bit_cast(v16bf, b1), o1);
  }
  #pragma unroll
  for (int r = 0; r < 8; ++r) {
    int m = wave * 16 + hi * 8 + r;
    if (m < 49) {
      size_t p = (size_t)token_pixel(b_, m, grid_mode);
      attn_out[p * 192 + head * 32 + nl]      = f2bf(o0[r]);
      attn_out[p * 192 + head * 32 + 16 + nl] = f2bf(o1[r]);
    }
  }
}

// ---------------- host launcher ----------------
extern "C" void kernel_launch(void* const* d_in, const int* in_sizes, int n_in,
                              void* d_out, int out_size, void* d_ws, size_t ws_size,
                              hipStream_t stream) {
  (void)in_sizes; (void)n_in; (void)out_size; (void)ws_size;

  const float* x      = (const float*)d_in[0];
  const float* bn0_g  = (const float*)d_in[1];
  const float* bn0_b  = (const float*)d_in[2];
  const float* bn0_m  = (const float*)d_in[3];
  const float* bn0_v  = (const float*)d_in[4];
  const float* pre_w  = (const float*)d_in[5];
  const float* pre_b  = (const float*)d_in[6];
  const float* dw_w   = (const float*)d_in[7];
  const float* dw_b   = (const float*)d_in[8];
  const float* dwbn_g = (const float*)d_in[9];
  const float* dwbn_b = (const float*)d_in[10];
  const float* dwbn_m = (const float*)d_in[11];
  const float* dwbn_v = (const float*)d_in[12];
  const float* pw_w   = (const float*)d_in[13];
  const float* pw_b   = (const float*)d_in[14];
  const float* pwbn_g = (const float*)d_in[15];
  const float* pwbn_b = (const float*)d_in[16];
  const float* pwbn_m = (const float*)d_in[17];
  const float* pwbn_v = (const float*)d_in[18];
  const float* se1_w  = (const float*)d_in[19];
  const float* se1_b  = (const float*)d_in[20];
  const float* se2_w  = (const float*)d_in[21];
  const float* se2_b  = (const float*)d_in[22];
  const float* post_w = (const float*)d_in[23];
  const float* post_b = (const float*)d_in[24];
  const float* sub_w  = (const float*)d_in[25];
  const float* sub_b  = (const float*)d_in[26];
  // wblk: 27..39, gblk: 40..52 (ln1_g, ln1_b, qkv_w, qkv_b, tab, proj_w, proj_b,
  //                             ln2_g, ln2_b, fc1_w, fc1_b, fc2_w, fc2_b)

  char* ws = (char*)d_ws;
  // Region1 (77.07MB): A0+h1 (MBConv)  ||  m1 (MLP hidden)
  unsigned short* A0  = (unsigned short*)(ws);
  unsigned short* h1  = (unsigned short*)(ws + (size_t)P0 * CINC * 2);
  unsigned short* m1  = (unsigned short*)(ws);
  const size_t R2 = (size_t)P1 * 768 * 2;                    // 77,070,336
  // Region2 (96.34MB): h2,h3,h3s,subA,bufSub  ||  tln,qkvB,attnO
  unsigned short* h2     = (unsigned short*)(ws + R2);
  unsigned short* h3     = (unsigned short*)(ws + R2 + 9633792u);
  unsigned short* h3s    = (unsigned short*)(ws + R2 + 28901376u);
  unsigned short* subA   = (unsigned short*)(ws + R2 + 48168960u);
  float*          bufSub = (float*)         (ws + R2 + 57802752u);
  unsigned short* tln    = (unsigned short*)(ws + R2);
  unsigned short* qkvB   = (unsigned short*)(ws + R2 + 19267584u);
  unsigned short* attnO  = (unsigned short*)(ws + R2 + 77070336u);
  const size_t R3 = R2 + 96337920u;
  float* bufX = (float*)(ws + R3);
  const size_t R4 = R3 + 38535168u;
  float* bufY = (float*)(ws + R4);
  size_t off = R4 + 38535168u;
  auto alloc = [&](size_t bytes) { void* p = ws + off; off += (bytes + 255u) & ~(size_t)255u; return p; };
  unsigned short* wPre  = (unsigned short*)alloc(128 * 96 * 2);   // padded 96 -> 128 rows
  unsigned short* wPw   = (unsigned short*)alloc(192 * 96 * 2);
  unsigned short* wPost = (unsigned short*)alloc(192 * 192 * 2);
  unsigned short* wSub  = (unsigned short*)alloc(192 * 96 * 2);
  unsigned short* wQkvW = (unsigned short*)alloc(576 * 192 * 2);
  unsigned short* wQkvG = (unsigned short*)alloc(576 * 192 * 2);
  unsigned short* wProjW= (unsigned short*)alloc(192 * 192 * 2);
  unsigned short* wProjG= (unsigned short*)alloc(192 * 192 * 2);
  unsigned short* wFc1W = (unsigned short*)alloc(768 * 192 * 2);
  unsigned short* wFc1G = (unsigned short*)alloc(768 * 192 * 2);
  unsigned short* wFc2W = (unsigned short*)alloc(192 * 768 * 2);
  unsigned short* wFc2G = (unsigned short*)alloc(192 * 768 * 2);
  float* scale0 = (float*)alloc(96 * 4);
  float* shift0 = (float*)alloc(96 * 4);
  float* sdw    = (float*)alloc(96 * 4);
  float* tdw    = (float*)alloc(96 * 4);
  float* spw    = (float*)alloc(192 * 4);
  float* tpw    = (float*)alloc(192 * 4);
  float* qscW   = (float*)alloc(576 * 4);
  float* qshW   = (float*)alloc(576 * 4);
  float* qscG   = (float*)alloc(576 * 4);
  float* qshG   = (float*)alloc(576 * 4);
  float* pool   = (float*)alloc(16 * 192 * 4);
  float* sesig  = (float*)alloc(16 * 192 * 4);
  float* separt = (float*)alloc(16 * 49 * 192 * 4);

  // ---- prep ----
  prep_params<<<1, 576, 0, stream>>>(bn0_g, bn0_b, bn0_m, bn0_v,
                                     dw_b, dwbn_g, dwbn_b, dwbn_m, dwbn_v,
                                     pw_b, pwbn_g, pwbn_b, pwbn_m, pwbn_v,
                                     (const float*)d_in[30], (const float*)d_in[43],
                                     scale0, shift0, sdw, tdw, spw, tpw,
                                     qscW, qshW, qscG, qshG);
  auto cvtw = [&](const float* src, unsigned short* dst, int n, int npad) {
    cvt_f2bf<<<(npad + 255) / 256, 256, 0, stream>>>(src, dst, n, npad);
  };
  cvtw(pre_w,  wPre,  96 * 96, 128 * 96);     // zero-pad rows 96..127
  cvtw(pw_w,   wPw,   192 * 96, 192 * 96);
  cvtw(post_w, wPost, 192 * 192, 192 * 192);
  cvtw(sub_w,  wSub,  192 * 96, 192 * 96);
  cvtw((const float*)d_in[29], wQkvW, 576 * 192, 576 * 192);
  cvtw((const float*)d_in[42], wQkvG, 576 * 192, 576 * 192);
  cvtw((const float*)d_in[32], wProjW, 192 * 192, 192 * 192);
  cvtw((const float*)d_in[45], wProjG, 192 * 192, 192 * 192);
  cvtw((const float*)d_in[36], wFc1W, 768 * 192, 768 * 192);
  cvtw((const float*)d_in[49], wFc1G, 768 * 192, 768 * 192);
  cvtw((const float*)d_in[38], wFc2W, 192 * 768, 192 * 768);
  cvtw((const float*)d_in[51], wFc2G, 192 * 768, 192 * 768);

  // ---- MBConv ----
  cvt0_kernel<<<(P0 * CINC + 255) / 256, 256, 0, stream>>>(x, scale0, shift0, A0);
  gemm_bf16<<<dim3(P0 / 128, 2), 256, 0, stream>>>(A0, wPre, nullptr, pre_b, nullptr, nullptr, h1,
                                                   P0, 96, 96, 0);
  dwconv_kernel<<<(P1 * CINC + 255) / 256, 256, 0, stream>>>(h1, dw_w, sdw, tdw, h2);
  gemm_bf16<<<dim3(P1 / 128, 3), 256, 0, stream>>>(h2, wPw, spw, tpw, nullptr, nullptr, h3,
                                                   P1, 192, 96, 1);
  se_pool_partial<<<dim3(16, 49), 192, 0, stream>>>(h3, separt);
  se_pool_final<<<16, 192, 0, stream>>>(separt, pool);
  se_fc<<<16, 192, 0, stream>>>(pool, se1_w, se1_b, se2_w, se2_b, sesig);
  se_scale<<<(P1 * COUTC + 255) / 256, 256, 0, stream>>>(h3, sesig, h3s);
  maxpool_cvt<<<(P1 * CINC + 255) / 256, 256, 0, stream>>>(x, subA);
  gemm_bf16<<<dim3(P1 / 128, 3), 256, 0, stream>>>(subA, wSub, nullptr, sub_b, nullptr, bufSub, nullptr,
                                                   P1, 192, 96, 0);
  gemm_bf16<<<dim3(P1 / 128, 3), 256, 0, stream>>>(h3s, wPost, nullptr, post_b, bufSub, bufX, nullptr,
                                                   P1, 192, 192, 0);

  // ---- transformer blocks (window then grid) ----
  auto run_block = [&](int base, unsigned short* wQ, unsigned short* wP,
                       unsigned short* w1, unsigned short* w2,
                       float* qsc, float* qsh, int gm) {
    const float* ln1_g = (const float*)d_in[base + 0];
    const float* ln1_b = (const float*)d_in[base + 1];
    const float* tab   = (const float*)d_in[base + 4];
    const float* proj_b= (const float*)d_in[base + 6];
    const float* ln2_g = (const float*)d_in[base + 7];
    const float* ln2_b = (const float*)d_in[base + 8];
    const float* fc1_b = (const float*)d_in[base + 10];
    const float* fc2_b = (const float*)d_in[base + 12];
    ln_kernel<<<P1 / 4, 128, 0, stream>>>(bufX, ln1_g, ln1_b, tln);
    gemm_bf16<<<dim3(P1 / 128, 9), 256, 0, stream>>>(tln, wQ, qsc, qsh, nullptr, nullptr, qkvB,
                                                     P1, 576, 192, 0);
    attn_core<<<dim3(1024, 6), 128, 0, stream>>>(qkvB, tab, attnO, gm);
    gemm_bf16<<<dim3(P1 / 128, 3), 256, 0, stream>>>(attnO, wP, nullptr, proj_b, bufX, bufY, nullptr,
                                                     P1, 192, 192, 0);
    ln_kernel<<<P1 / 4, 128, 0, stream>>>(bufY, ln2_g, ln2_b, tln);
    gemm_bf16<<<dim3(P1 / 128, 12), 256, 0, stream>>>(tln, w1, nullptr, fc1_b, nullptr, nullptr, m1,
                                                      P1, 768, 192, 1);
    gemm_bf16<<<dim3(P1 / 128, 3), 256, 0, stream>>>(m1, w2, nullptr, fc2_b, bufY, bufX, nullptr,
                                                     P1, 192, 768, 0);
  };
  run_block(27, wQkvW, wProjW, wFc1W, wFc2W, qscW, qshW, 0);   // window attention
  run_block(40, wQkvG, wProjG, wFc1G, wFc2G, qscG, qshG, 1);   // grid attention

  out_transpose<<<(P1 * COUTC + 255) / 256, 256, 0, stream>>>(bufX, (float*)d_out);
}
